// SSDKernel_70738111365258
// MI455X (gfx1250) — matrix operand order repaired and checked by reading, not compiled, not run on hardware
//
#include <hip/hip_runtime.h>
#include <hip/hip_bf16.h>

// ---------------------------------------------------------------------------
// Mamba-2 style SSD block for MI455X (gfx1250, wave32, WMMA + TDM).
// B=4, L=4096, D=512, N=64, CHUNK=64.
// ---------------------------------------------------------------------------

typedef __bf16 bf16;
typedef __attribute__((ext_vector_type(16))) __bf16 v16bf;
typedef __attribute__((ext_vector_type(8)))  __bf16 v8bf;
typedef __attribute__((ext_vector_type(8)))  float  v8f;
typedef unsigned int u32x4 __attribute__((ext_vector_type(4)));
typedef int          i32x4 __attribute__((ext_vector_type(4)));
typedef int          i32x8 __attribute__((ext_vector_type(8)));

#define BL   16384   // B*L rows
#define DIMD 512
#define NS   64      // state dim
#define PCH  64      // chunk length
#define NCH  64      // number of chunks
#define NBAT 4
#define SEQL 4096

#if defined(__has_builtin)
#if __has_builtin(__builtin_amdgcn_tensor_load_to_lds) && __has_builtin(__builtin_amdgcn_s_wait_tensorcnt)
#define USE_TDM 1
#endif
#endif

__device__ __forceinline__ v16bf combine16(v8bf lo, v8bf hi8) {
  v16bf r;
#pragma unroll
  for (int e = 0; e < 8; ++e) { r[e] = lo[e]; r[e + 8] = hi8[e]; }
  return r;
}

#ifdef USE_TDM
// ---------------------------------------------------------------------------
// Tensor Data Mover: 2D tile load (2-byte elements) global -> LDS.
// Tile = tile0 x tile1 elements; tensor extents (from tile origin) tdim0/tdim1
// for OOB zero-fill; row stride stride0 (elements).  LDS padding: 4 DWORDs
// every 16 DWORDs => LDS row stride = 40 halves (matches fragment readers).
// D# layout per CDNA5 ISA 8.3/8.4. Groups 2/3 zero (2D tensor).
// clang-23 toolchain: 6-arg builtin (g0 v4u, g1 v8i, v4i, v4i, v8i, cpol).
// ---------------------------------------------------------------------------
__device__ __forceinline__ void tdm_load_tile_2d(
    unsigned lds_byte_off, const void* gptr,
    unsigned tdim0, unsigned tdim1,
    unsigned tile0, unsigned tile1,
    unsigned long long stride0_elems) {
  unsigned long long ga = (unsigned long long)gptr;
  u32x4 g0;
  g0.x = 1u;                                                   // count=1 (valid)
  g0.y = lds_byte_off;                                         // lds_addr
  g0.z = (unsigned)ga;                                         // global_addr[31:0]
  g0.w = (unsigned)((ga >> 32) & 0x01FFFFFFu) | 0x80000000u;   // addr[56:32] | type=2
  i32x8 g1;
  g1[0] = (int)((1u << 16) | (1u << 20) | (3u << 22) | (3u << 25));
  //            data_size=2B  pad_en     interval=16dw  amount=4dw
  g1[1] = (int)(tdim0 << 16);                                  // tensor_dim0[15:0]
  g1[2] = (int)((tdim0 >> 16) | (tdim1 << 16));                // td0 hi | td1 lo
  g1[3] = (int)((tdim1 >> 16) | (tile0 << 16));                // td1 hi | tile_dim0
  g1[4] = (int)(tile1 & 0xFFFFu);                              // tile_dim1 (tile_dim2=0)
  g1[5] = (int)(unsigned)stride0_elems;                        // stride0[31:0]
  g1[6] = (int)((unsigned)(stride0_elems >> 32) & 0xFFFFu);    // stride0[47:32]
  g1[7] = 0;
  i32x4 gz4 = {0, 0, 0, 0};
  i32x8 gz8 = {0, 0, 0, 0, 0, 0, 0, 0};
  __builtin_amdgcn_tensor_load_to_lds(g0, g1, gz4, gz4, gz8, 0);
}

__device__ __forceinline__ unsigned lds_off32(const void* p) {
  return (unsigned)(unsigned long long)p;   // low 32 bits = LDS offset
}
#endif

// ---------------------------------------------------------------------------
// f32 -> bf16 convert (strided source rows)
// ---------------------------------------------------------------------------
__global__ void k_cvt_bf16(const float* __restrict__ src, bf16* __restrict__ dst,
                           int rows, int cols, int src_stride) {
  int idx = blockIdx.x * blockDim.x + threadIdx.x;
  if (idx >= rows * cols) return;
  int r = idx / cols, c = idx % cols;
  dst[idx] = (bf16)src[(size_t)r * src_stride + c];
}

// f32 (K x N) -> bf16 transposed (N x K): dst[n*K + k] = src[k*N + n]
__global__ void k_cvt_t_bf16(const float* __restrict__ src, bf16* __restrict__ dst,
                             int K, int N) {
  int idx = blockIdx.x * blockDim.x + threadIdx.x;
  if (idx >= K * N) return;
  int n = idx / K, k = idx % K;
  dst[idx] = (bf16)src[(size_t)k * N + n];
}

// ---------------------------------------------------------------------------
// Tiled bf16 WMMA GEMM with TDM-fed double-buffered LDS.
//   C[M,N] = A[M,K](row-major, lda) * BT[N,K](row-major)^T + bias
// Block tile 128x128, 4 waves, wave tile 64x64 = 4x4 WMMA tiles. K-step 32.
// LDS tiles are [row][k] with row stride 40 halves (TDM pad reproduces this).
// ---------------------------------------------------------------------------
#define LSTR 40   // 32 data + 8 pad halves per LDS row (16B aligned)

__global__ __launch_bounds__(128)
void k_gemm_bf16(const bf16* __restrict__ A, const bf16* __restrict__ BT,
                 const float* __restrict__ bias, float* __restrict__ Co,
                 int M, int N, int K, int lda, int ldc) {
  __shared__ bf16 sA[2][128 * LSTR];
  __shared__ bf16 sB[2][128 * LSTR];

  const int tid  = threadIdx.x;
  const int lane = tid & 31;
  const int wave = tid >> 5;
  const int wm   = (wave >> 1) * 64;
  const int wn   = (wave & 1) * 64;
  const int m0   = blockIdx.y * 128;
  const int n0   = blockIdx.x * 128;
  const int c16  = lane & 15;
  const int hi   = lane >> 4;

  v8f acc[4][4] = {};
  const int nk = K / 32;

#ifdef USE_TDM
  if (wave == 0) {  // one wave drives the DMA engine
    tdm_load_tile_2d(lds_off32(&sA[0][0]), A + (size_t)m0 * lda,
                     (unsigned)K, (unsigned)(M - m0), 32, 128,
                     (unsigned long long)lda);
    tdm_load_tile_2d(lds_off32(&sB[0][0]), BT + (size_t)n0 * K,
                     (unsigned)K, (unsigned)(N - n0), 32, 128,
                     (unsigned long long)K);
  }
#endif

  for (int kb = 0; kb < nk; ++kb) {
    const int cur = kb & 1;
    const int k0  = kb * 32;

#ifdef USE_TDM
    if (wave == 0) {
      if (kb + 1 < nk) {
        int k0n = k0 + 32;
        tdm_load_tile_2d(lds_off32(&sA[cur ^ 1][0]), A + (size_t)m0 * lda + k0n,
                         (unsigned)(K - k0n), (unsigned)(M - m0), 32, 128,
                         (unsigned long long)lda);
        tdm_load_tile_2d(lds_off32(&sB[cur ^ 1][0]), BT + (size_t)n0 * K + k0n,
                         (unsigned)(K - k0n), (unsigned)(N - n0), 32, 128,
                         (unsigned long long)K);
        __builtin_amdgcn_s_wait_tensorcnt(2);  // current buffer's 2 loads done
      } else {
        __builtin_amdgcn_s_wait_tensorcnt(0);
      }
    }
    __syncthreads();
#else
    // Fallback: manual staging (both operands are [row][k], no transpose).
    {
      const bf16* sa = A + (size_t)(m0 + tid) * lda + k0;
      bf16* dp = &sA[cur][tid * LSTR];
#pragma unroll
      for (int i = 0; i < 4; ++i) *(v8bf*)(dp + i * 8) = *(const v8bf*)(sa + i * 8);
      bf16* dq = &sB[cur][tid * LSTR];
      if (n0 + tid < N) {
        const bf16* sb = BT + (size_t)(n0 + tid) * K + k0;
#pragma unroll
        for (int i = 0; i < 4; ++i) *(v8bf*)(dq + i * 8) = *(const v8bf*)(sb + i * 8);
      } else {
        v8bf z = {};
#pragma unroll
        for (int i = 0; i < 4; ++i) *(v8bf*)(dq + i * 8) = z;
      }
    }
    __syncthreads();
#endif

    // --- fragments ---
    v16bf af[4], bfg[4];
#pragma unroll
    for (int mt = 0; mt < 4; ++mt) {
      const bf16* p = &sA[cur][(wm + mt * 16 + c16) * LSTR + hi * 8];
      af[mt] = combine16(*(const v8bf*)p, *(const v8bf*)(p + 16));
    }
#pragma unroll
    for (int nt = 0; nt < 4; ++nt) {
      const bf16* p = &sB[cur][(wn + nt * 16 + c16) * LSTR + hi * 8];
      bfg[nt] = combine16(*(const v8bf*)p, *(const v8bf*)(p + 16));
    }
    // --- 16 WMMA per wave per K-step ---
#pragma unroll
    for (int mt = 0; mt < 4; ++mt)
#pragma unroll
      for (int nt = 0; nt < 4; ++nt)
        acc[mt][nt] = __builtin_amdgcn_wmma_f32_16x16x32_bf16(
            false, af[mt], false, bfg[nt], (short)0, acc[mt][nt], false, false);
    __syncthreads();   // protect LDS buffer before DMA/staging overwrites it
  }

  // --- epilogue ---
#pragma unroll
  for (int nt = 0; nt < 4; ++nt) {
    int col = n0 + wn + nt * 16 + c16;
    if (col >= N) continue;
    float bv = bias ? bias[col] : 0.0f;
#pragma unroll
    for (int mt = 0; mt < 4; ++mt) {
#pragma unroll
      for (int r = 0; r < 8; ++r) {
        int rowg = m0 + wm + mt * 16 + r + 8 * hi;
        Co[(size_t)rowg * ldc + col] = acc[mt][nt][r] + bv;
      }
    }
  }
}

// ---------------------------------------------------------------------------
// Row-wise rmsnorm over 64 cols -> bf16. 8 waves/block, 1 row per wave.
// ---------------------------------------------------------------------------
__global__ __launch_bounds__(256)
void k_rms64(const float* __restrict__ in, const float* __restrict__ g,
             bf16* __restrict__ outp, int rows) {
  int wave = threadIdx.x >> 5, lane = threadIdx.x & 31;
  int row = blockIdx.x * 8 + wave;
  if (row >= rows) return;
  const float* p = in + (size_t)row * 64;
  float v0 = p[lane], v1 = p[lane + 32];
  float ss = v0 * v0 + v1 * v1;
#pragma unroll
  for (int off = 16; off > 0; off >>= 1) ss += __shfl_xor(ss, off, 32);
  float rn = rsqrtf(ss * (1.0f / 64.0f) + 1e-6f);
  outp[(size_t)row * 64 + lane]      = (bf16)(v0 * rn * g[lane]);
  outp[(size_t)row * 64 + lane + 32] = (bf16)(v1 * rn * g[lane + 32]);
}

// ---------------------------------------------------------------------------
// dt activation: la = max(clip(softplus(dtraw)*A, -20, 0), log(1e-8)), in place.
// ---------------------------------------------------------------------------
__global__ void k_dt_la(float* __restrict__ dtraw, const float* __restrict__ A_log,
                        int total) {
  int i = blockIdx.x * blockDim.x + threadIdx.x;
  if (i >= total) return;
  int d = i & (DIMD - 1);
  float v  = dtraw[i];
  float sp = (v > 20.0f) ? v : log1pf(__expf(v));
  float dt = fminf(fmaxf(sp, 1e-4f), 5.0f);
  float Al = fminf(fmaxf(A_log[d], -20.0f), 2.0f);
  float la = fminf(fmaxf(dt * (-__expf(Al)), -20.0f), 0.0f);
  dtraw[i] = fmaxf(la, -18.420681f);
}

// ---------------------------------------------------------------------------
// G = tril(C_chunk @ B_chunk^T) per (chunk,batch): batched 64x64x64 WMMA.
// ---------------------------------------------------------------------------
__global__ __launch_bounds__(128)
void k_gmat(const bf16* __restrict__ Cm, const bf16* __restrict__ Bm,
            float* __restrict__ G) {
  int cb = blockIdx.x;             // ci*NBAT + b
  int ci = cb >> 2, b = cb & 3;
  int lane = threadIdx.x & 31, wave = threadIdx.x >> 5;
  int c16 = lane & 15, hi = lane >> 4;
  size_t rowbase = (size_t)b * SEQL + (size_t)ci * PCH;

  v8f acc[4] = {};
#pragma unroll
  for (int k0 = 0; k0 < NS; k0 += 32) {
    const bf16* pa = Cm + (rowbase + wave * 16 + c16) * NS + k0 + hi * 8;
    v16bf a = combine16(*(const v8bf*)pa, *(const v8bf*)(pa + 16));
#pragma unroll
    for (int jt = 0; jt < 4; ++jt) {
      const bf16* pb = Bm + (rowbase + jt * 16 + c16) * NS + k0 + hi * 8;
      v16bf bb = combine16(*(const v8bf*)pb, *(const v8bf*)(pb + 16));
      acc[jt] = __builtin_amdgcn_wmma_f32_16x16x32_bf16(
          false, a, false, bb, (short)0, acc[jt], false, false);
    }
  }
#pragma unroll
  for (int jt = 0; jt < 4; ++jt) {
    int j = jt * 16 + c16;
#pragma unroll
    for (int r = 0; r < 8; ++r) {
      int i = wave * 16 + r + 8 * hi;
      G[((size_t)cb * 64 + i) * 64 + j] = (i >= j) ? acc[jt][r] : 0.0f;
    }
  }
}

// ---------------------------------------------------------------------------
// Chunk-sequential SSD scan. Grid = B * (D/32). thread = (pg=wave, d=lane).
// ---------------------------------------------------------------------------
#define DT 32
__global__ __launch_bounds__(256)
void k_scan(const float* __restrict__ xbar, int xstride,
            const float* __restrict__ la,
            const bf16* __restrict__ Bm, const bf16* __restrict__ Cm,
            const float* __restrict__ G, float* __restrict__ Y) {
  __shared__ float sLa[PCH][DT + 1];
  __shared__ float sX [PCH][DT + 1];
  __shared__ float sSt[NS ][DT + 1];
  __shared__ float sSum[8][DT + 1];

  const int d  = threadIdx.x & 31;
  const int pg = threadIdx.x >> 5;
  const int b   = blockIdx.x >> 4;
  const int dt0 = (blockIdx.x & 15) * DT;

  for (int n = pg; n < NS; n += 8) sSt[n][d] = 0.0f;
  __syncthreads();

  for (int ci = 0; ci < NCH; ++ci) {
    size_t rowbase = (size_t)b * SEQL + (size_t)ci * PCH;
    if (ci + 1 < NCH) {
      size_t nb = rowbase + PCH;
      __builtin_prefetch(la   + (nb + pg * 8) * DIMD   + dt0 + d, 0, 0);
      __builtin_prefetch(xbar + (nb + pg * 8) * xstride + dt0 + d, 0, 0);
    }

    // 1) load la & x slice, cumsum la over p
    float myla[8];
#pragma unroll
    for (int q = 0; q < 8; ++q) {
      int p = pg * 8 + q;
      myla[q]  = la  [(rowbase + p) * DIMD    + dt0 + d];
      sX[p][d] = xbar[(rowbase + p) * xstride + dt0 + d];
    }
#pragma unroll
    for (int q = 1; q < 8; ++q) myla[q] += myla[q - 1];
    sSum[pg][d] = myla[7];
    __syncthreads();
    float base = 0.0f;
    for (int g2 = 0; g2 < pg; ++g2) base += sSum[g2][d];
#pragma unroll
    for (int q = 0; q < 8; ++q) sLa[pg * 8 + q][d] = myla[q] + base;
    __syncthreads();

    const float laN = sLa[PCH - 1][d];

    // 2) y = y_inter + y_intra for this thread's 8 rows
    for (int q = 0; q < 8; ++q) {
      int i = pg * 8 + q;
      float lci = sLa[i][d];
      float yin = 0.0f;
      const bf16* crow = Cm + (rowbase + i) * NS;
      for (int n = 0; n < NS; ++n) yin += (float)crow[n] * sSt[n][d];
      yin *= __expf(lci);
      float yia = 0.0f;
      const float* grow = G + ((size_t)(ci * NBAT + b) * 64 + i) * 64;
      for (int j = 0; j <= i; ++j)
        yia += grow[j] * __expf(lci - sLa[j][d]) * sX[j][d];
      float y = yin + yia;
      if (!__builtin_isfinite(y)) y = 0.0f;
      Y[(rowbase + i) * DIMD + dt0 + d] = y;
    }
    __syncthreads();

    // 3) state update: thread owns n = pg*8 .. pg*8+7 for its d column
    float atot = __expf(laN);
    float accn[8] = {};
    for (int p = 0; p < PCH; ++p) {
      float xs = sX[p][d] * __expf(laN - sLa[p][d]);
      const bf16* brow = Bm + (rowbase + p) * NS + pg * 8;
#pragma unroll
      for (int q = 0; q < 8; ++q) accn[q] += (float)brow[q] * xs;
    }
#pragma unroll
    for (int q = 0; q < 8; ++q) {
      int n = pg * 8 + q;
      float nsv = sSt[n][d] * atot + accn[q];
      if (!__builtin_isfinite(nsv)) nsv = 0.0f;
      sSt[n][d] = nsv;
    }
    __syncthreads();
  }
}

// ---------------------------------------------------------------------------
// Output gate: y = rmsnorm(y_ssd + D_skip*x_bar) * silu(z)  -> bf16
// ---------------------------------------------------------------------------
__global__ __launch_bounds__(256)
void k_post(const float* __restrict__ Yssd, const float* __restrict__ xp,
            const float* __restrict__ Dskip, const float* __restrict__ gout,
            bf16* __restrict__ outp) {
  __shared__ float red[8];
  int r = blockIdx.x, tid = threadIdx.x;
  const float* xb = xp + (size_t)r * 1024;
  const float* zp = xb + 512;
  float v[2], ss = 0.0f;
#pragma unroll
  for (int q = 0; q < 2; ++q) {
    int dch = tid + q * 256;
    float y = Yssd[(size_t)r * DIMD + dch];
    if (!__builtin_isfinite(y)) y = 0.0f;
    y += Dskip[dch] * xb[dch];
    v[q] = y; ss += y * y;
  }
#pragma unroll
  for (int off = 16; off > 0; off >>= 1) ss += __shfl_xor(ss, off, 32);
  if ((tid & 31) == 0) red[tid >> 5] = ss;
  __syncthreads();
  float tot = 0.0f;
#pragma unroll
  for (int w = 0; w < 8; ++w) tot += red[w];
  float rn = rsqrtf(tot * (1.0f / 512.0f) + 1e-6f);
#pragma unroll
  for (int q = 0; q < 2; ++q) {
    int dch = tid + q * 256;
    float z = zp[dch];
    float silu = z / (1.0f + __expf(-z));
    outp[(size_t)r * DIMD + dch] = (bf16)(v[q] * rn * gout[dch] * silu);
  }
}

// ---------------------------------------------------------------------------
extern "C" void kernel_launch(void* const* d_in, const int* in_sizes, int n_in,
                              void* d_out, int out_size, void* d_ws, size_t ws_size,
                              hipStream_t stream) {
  const float* x         = (const float*)d_in[0];
  const float* A_log     = (const float*)d_in[1];
  const float* dt_bias   = (const float*)d_in[2];
  const float* D_skip    = (const float*)d_in[3];
  const float* W_xproj   = (const float*)d_in[4];
  const float* b_xproj   = (const float*)d_in[5];
  const float* W_B       = (const float*)d_in[6];
  const float* W_C       = (const float*)d_in[7];
  const float* W_dt      = (const float*)d_in[8];
  const float* g_Bnorm   = (const float*)d_in[9];
  const float* g_Cnorm   = (const float*)d_in[10];
  const float* g_outnorm = (const float*)d_in[11];
  const float* W_out     = (const float*)d_in[12];
  const float* b_out     = (const float*)d_in[13];
  float* out = (float*)d_out;

  char* ws = (char*)d_ws;
  size_t off = 0;
  auto walloc = [&](size_t bytes) -> void* {
    void* p = ws + off;
    off += (bytes + 255) & ~(size_t)255;
    return p;
  };

  bf16* xbf     = (bf16*)walloc((size_t)BL * DIMD * 2);
  bf16* wxpT    = (bf16*)walloc((size_t)1024 * DIMD * 2);   // (2D x D)
  bf16* wbT     = (bf16*)walloc((size_t)NS * DIMD * 2);     // (N x D)
  bf16* wcT     = (bf16*)walloc((size_t)NS * DIMD * 2);
  bf16* wdtT    = (bf16*)walloc((size_t)DIMD * DIMD * 2);
  bf16* woutT   = (bf16*)walloc((size_t)DIMD * DIMD * 2);
  float* xp     = (float*)walloc((size_t)BL * 1024 * 4);    // [x_bar | z]
  bf16* xbar_bf = (bf16*)walloc((size_t)BL * DIMD * 2);
  float* braw   = (float*)walloc((size_t)BL * NS * 4);
  float* craw   = (float*)walloc((size_t)BL * NS * 4);
  bf16* bm_bf   = (bf16*)walloc((size_t)BL * NS * 2);
  bf16* cm_bf   = (bf16*)walloc((size_t)BL * NS * 2);
  float* dtla   = (float*)walloc((size_t)BL * DIMD * 4);
  float* Gm     = (float*)walloc((size_t)NCH * NBAT * 64 * 64 * 4);
  float* yssd   = (float*)walloc((size_t)BL * DIMD * 4);
  bf16* yfin_bf = (bf16*)walloc((size_t)BL * DIMD * 2);

  const int T = 256;
  // converts: activations row-major, weights transposed (N-major) for TDM tiles
  k_cvt_bf16<<<(BL * DIMD + T - 1) / T, T, 0, stream>>>(x, xbf, BL, DIMD, DIMD);
  k_cvt_t_bf16<<<(DIMD * 1024 + T - 1) / T, T, 0, stream>>>(W_xproj, wxpT, DIMD, 1024);
  k_cvt_t_bf16<<<(DIMD * NS + T - 1) / T, T, 0, stream>>>(W_B, wbT, DIMD, NS);
  k_cvt_t_bf16<<<(DIMD * NS + T - 1) / T, T, 0, stream>>>(W_C, wcT, DIMD, NS);
  k_cvt_t_bf16<<<(DIMD * DIMD + T - 1) / T, T, 0, stream>>>(W_dt, wdtT, DIMD, DIMD);
  k_cvt_t_bf16<<<(DIMD * DIMD + T - 1) / T, T, 0, stream>>>(W_out, woutT, DIMD, DIMD);

  // xp = x @ W_xproj + b_xproj
  k_gemm_bf16<<<dim3(1024 / 128, BL / 128), 128, 0, stream>>>(
      xbf, wxpT, b_xproj, xp, BL, 1024, DIMD, DIMD, 1024);

  // x_bar (strided view of xp) -> bf16
  k_cvt_bf16<<<(BL * DIMD + T - 1) / T, T, 0, stream>>>(xp, xbar_bf, BL, DIMD, 1024);

  // B / C / dt projections
  k_gemm_bf16<<<dim3(1, BL / 128), 128, 0, stream>>>(
      xbar_bf, wbT, nullptr, braw, BL, NS, DIMD, DIMD, NS);
  k_gemm_bf16<<<dim3(1, BL / 128), 128, 0, stream>>>(
      xbar_bf, wcT, nullptr, craw, BL, NS, DIMD, DIMD, NS);
  k_gemm_bf16<<<dim3(DIMD / 128, BL / 128), 128, 0, stream>>>(
      xbar_bf, wdtT, dt_bias, dtla, BL, DIMD, DIMD, DIMD, DIMD);

  // epilogues
  k_rms64<<<BL / 8, 256, 0, stream>>>(braw, g_Bnorm, bm_bf, BL);
  k_rms64<<<BL / 8, 256, 0, stream>>>(craw, g_Cnorm, cm_bf, BL);
  k_dt_la<<<(BL * DIMD + T - 1) / T, T, 0, stream>>>(dtla, A_log, BL * DIMD);

  // G = tril(C @ B^T) per (chunk,batch)
  k_gmat<<<NCH * NBAT, 128, 0, stream>>>(cm_bf, bm_bf, Gm);

  // chunk-sequential scan
  k_scan<<<NBAT * (DIMD / DT), 256, 0, stream>>>(
      xp, 1024, dtla, bm_bf, cm_bf, Gm, yssd);

  // gated output norm
  k_post<<<BL, 256, 0, stream>>>(yssd, xp, D_skip, g_outnorm, yfin_bf);

  // out = y @ W_out + b_out
  k_gemm_bf16<<<dim3(DIMD / 128, BL / 128), 128, 0, stream>>>(
      yfin_bf, woutT, b_out, out, BL, DIMD, DIMD, DIMD, DIMD);
}
